// TopM_MHSA_65635690218075
// MI455X (gfx1250) — matrix operand order repaired and checked by reading, not compiled, not run on hardware
//
#include <hip/hip_runtime.h>

#define LNUM 2
#define BB 4
#define NN 1024
#define DD 512
#define HH 8
#define DHH 64
#define FF 2048
#define MTOP 64
#define BN_TOT (BB * NN)   /* 4096 */
#define EPSV 1e-5f
#define QSCALE 0.125f      /* 64^-0.5 */

typedef __attribute__((ext_vector_type(16))) __bf16 v16bf;
typedef __attribute__((ext_vector_type(8)))  float  v8f;

union AFrag { v16bf v; unsigned short s[16]; uint4 q[2]; };
union CFrag { v8f  v; float f[8]; };

// ---------- small helpers ----------
__device__ __forceinline__ unsigned short f32_to_bf16_bits(float f) {
    unsigned u = __builtin_bit_cast(unsigned, f);
    u += 0x7FFFu + ((u >> 16) & 1u);           // round-to-nearest-even
    return (unsigned short)(u >> 16);
}
__device__ __forceinline__ unsigned sortkey(float f) {
    unsigned u = __builtin_bit_cast(unsigned, f);
    return (u & 0x80000000u) ? ~u : (u | 0x80000000u);
}
__device__ __forceinline__ float wred_sum(float v) {
    #pragma unroll
    for (int m = 16; m >= 1; m >>= 1) v += __shfl_xor(v, m, 32);
    return v;
}
__device__ __forceinline__ float wred_max(float v) {
    #pragma unroll
    for (int m = 16; m >= 1; m >>= 1) v = fmaxf(v, __shfl_xor(v, m, 32));
    return v;
}
__device__ __forceinline__ int wred_sumi(int v) {
    #pragma unroll
    for (int m = 16; m >= 1; m >>= 1) v += __shfl_xor(v, m, 32);
    return v;
}
// low 32 bits of a generic pointer to LDS == wave-relative LDS byte offset
__device__ __forceinline__ unsigned lds_addr32(const void* p) {
    return (unsigned)(unsigned long long)p;
}

// ---------- utility kernels ----------
__global__ void cvt_bf16_kernel(const float* __restrict__ in, unsigned short* __restrict__ out, int n) {
    for (int i = blockIdx.x * blockDim.x + threadIdx.x; i < n; i += gridDim.x * blockDim.x)
        out[i] = f32_to_bf16_bits(in[i]);
}
__global__ void copy_f32_kernel(const float* __restrict__ in, float* __restrict__ out, int n) {
    for (int i = blockIdx.x * blockDim.x + threadIdx.x; i < n; i += gridDim.x * blockDim.x)
        out[i] = in[i];
}

// ---------- LayerNorm -> bf16 (one 256-thread block per token row, D=512) ----------
__global__ __launch_bounds__(256) void ln_bf16_kernel(
    const float* __restrict__ x, const float* __restrict__ g, const float* __restrict__ bta,
    unsigned short* __restrict__ out)
{
    __shared__ float ssum[8], ssq[8];
    const int row = blockIdx.x, tid = threadIdx.x;
    const float* xr = x + (long)row * DD;
    float v0 = xr[tid], v1 = xr[tid + 256];
    float s = wred_sum(v0 + v1);
    float q = wred_sum(v0 * v0 + v1 * v1);
    const int wid = tid >> 5, lane = tid & 31;
    if (lane == 0) { ssum[wid] = s; ssq[wid] = q; }
    __syncthreads();
    float ts = 0.f, tq = 0.f;
    #pragma unroll
    for (int i = 0; i < 8; ++i) { ts += ssum[i]; tq += ssq[i]; }
    const float mean = ts * (1.f / DD);
    const float var  = tq * (1.f / DD) - mean * mean;
    const float rstd = rsqrtf(var + EPSV);
    out[(long)row * DD + tid]       = f32_to_bf16_bits((v0 - mean) * rstd * g[tid]       + bta[tid]);
    out[(long)row * DD + tid + 256] = f32_to_bf16_bits((v1 - mean) * rstd * g[tid + 256] + bta[tid + 256]);
}

// ---------- bf16 WMMA GEMM: C[M,N] = act((A@B + bias)*alpha) (+resid) ----------
// Block tile 128x64, 8 waves, K-step 32. A tile staged via async copy to LDS
// (double-buffered); B tile staged transposed so fragments are b128 DS loads.
#define BT_STR 40   /* padded k-stride of transposed B tile */
__global__ __launch_bounds__(256) void gemm_bf16_kernel(
    const unsigned short* __restrict__ A, const unsigned short* __restrict__ Bm,
    const float* __restrict__ bias, const float* __restrict__ resid,
    float* __restrict__ outF, unsigned short* __restrict__ outBF,
    int Mdim, int Ndim, int Kdim, float alpha, int doGelu)
{
    __shared__ unsigned short sA[2][128 * 32];
    __shared__ unsigned short sBt[2][64 * BT_STR];
    const int tid = threadIdx.x;
    const int wid = tid >> 5, lane = tid & 31;
    const int m16 = lane & 15, hf = lane >> 4;
    const long bm = (long)blockIdx.y * 128;
    const long bn = (long)blockIdx.x * 64;

    const int arow = tid >> 2;            // 0..63
    const int acol = (tid & 3) * 8;       // 0,8,16,24
    const int brow = tid >> 3;            // 0..31   (k within tile)
    const int bcol = (tid & 7) * 8;       // 0..56   (n within tile)

    auto stageA = [&](int buf, int k0) {  // 128x32 tile: 2 async b128 copies/thread
        unsigned l0 = lds_addr32(&sA[buf][arow * 32 + acol]);
        unsigned l1 = lds_addr32(&sA[buf][(arow + 64) * 32 + acol]);
        const unsigned short* g0 = A + (bm + arow)      * (long)Kdim + k0 + acol;
        const unsigned short* g1 = A + (bm + arow + 64) * (long)Kdim + k0 + acol;
        asm volatile("global_load_async_to_lds_b128 %0, %1, off" :: "v"(l0), "v"(g0) : "memory");
        asm volatile("global_load_async_to_lds_b128 %0, %1, off" :: "v"(l1), "v"(g1) : "memory");
    };
    auto stageB = [&](int buf, int k0) {  // 32x64 tile, stored transposed [n][k]
        uint4 d = *(const uint4*)(Bm + (long)(k0 + brow) * Ndim + bn + bcol);
        const unsigned short* p = (const unsigned short*)&d;
        #pragma unroll
        for (int j = 0; j < 8; ++j)
            sBt[buf][(bcol + j) * BT_STR + brow] = p[j];
    };

    CFrag acc[4];
    #pragma unroll
    for (int t = 0; t < 4; ++t)
        #pragma unroll
        for (int r = 0; r < 8; ++r) acc[t].f[r] = 0.f;

    stageA(0, 0);
    stageB(0, 0);
    asm volatile("s_wait_asynccnt 0x0" ::: "memory");
    __syncthreads();

    int cur = 0;
    for (int k0 = 0; k0 < Kdim; k0 += 32) {
        const int nxt = cur ^ 1;
        if (k0 + 32 < Kdim) {             // prefetch next tile while computing
            stageA(nxt, k0 + 32);
            stageB(nxt, k0 + 32);
            __builtin_prefetch(Bm + (long)(k0 + 64 + brow) * Ndim + bn + bcol, 0, 1);
        }
        AFrag a;                          // 2x ds_load_b128
        a.q[0] = *(const uint4*)(&sA[cur][(wid * 16 + m16) * 32 + hf * 8]);
        a.q[1] = *(const uint4*)(&sA[cur][(wid * 16 + m16) * 32 + 16 + hf * 8]);
        #pragma unroll
        for (int nt = 0; nt < 4; ++nt) {
            AFrag bfr;                    // 2x ds_load_b128 from transposed tile
            bfr.q[0] = *(const uint4*)(&sBt[cur][(nt * 16 + m16) * BT_STR + hf * 8]);
            bfr.q[1] = *(const uint4*)(&sBt[cur][(nt * 16 + m16) * BT_STR + 16 + hf * 8]);
            acc[nt].v = __builtin_amdgcn_wmma_f32_16x16x32_bf16(
                false, a.v, false, bfr.v, (short)0, acc[nt].v, false, false);
        }
        asm volatile("s_wait_asynccnt 0x0" ::: "memory");
        __syncthreads();
        cur = nxt;
    }

    #pragma unroll
    for (int nt = 0; nt < 4; ++nt) {
        const long ncol = bn + nt * 16 + m16;
        const float bv = bias ? bias[ncol] : 0.f;
        #pragma unroll
        for (int r = 0; r < 8; ++r) {
            const long mrow = bm + wid * 16 + hf * 8 + r;
            float v = (acc[nt].f[r] + bv) * alpha;
            if (doGelu) v = 0.5f * v * (1.f + erff(v * 0.70710678118f));
            if (resid)  v += resid[mrow * Ndim + ncol];
            if (outF)  outF[mrow * Ndim + ncol]  = v;
            if (outBF) outBF[mrow * Ndim + ncol] = f32_to_bf16_bits(v);
        }
    }
}

// ---------- per-row full softmax + top-M masked softmax blend ----------
__device__ __forceinline__ void process_row(const float* rv, unsigned short* probRow,
                                            int lane, float wn0, float wn1)
{
    unsigned key[32];
    float mx = -3.0e38f;
    #pragma unroll
    for (int j = 0; j < 32; ++j) { key[j] = sortkey(rv[j]); mx = fmaxf(mx, rv[j]); }
    mx = wred_max(mx);
    float sfull = 0.f;
    #pragma unroll
    for (int j = 0; j < 32; ++j) sfull += __expf(rv[j] - mx);
    sfull = wred_sum(sfull);
    unsigned lo = 0u;                      // bitwise binary search for M-th largest
    for (int bit = 31; bit >= 0; --bit) {
        const unsigned t = lo | (1u << bit);
        int c = 0;
        #pragma unroll
        for (int j = 0; j < 32; ++j) c += (key[j] >= t) ? 1 : 0;
        c = wred_sumi(c);
        if (c >= MTOP) lo = t;
    }
    float stop = 0.f;
    #pragma unroll
    for (int j = 0; j < 32; ++j) stop += (key[j] >= lo) ? __expf(rv[j] - mx) : 0.f;
    stop = wred_sum(stop);
    const float cf = wn0 / sfull, ct = wn1 / stop;
    #pragma unroll
    for (int j = 0; j < 32; ++j) {
        const float e = __expf(rv[j] - mx);
        const float p = e * cf + ((key[j] >= lo) ? e * ct : 0.f);
        probRow[lane + 32 * j] = f32_to_bf16_bits(p);
    }
}

// ---------- attention: one block = (b,h) x 16 query rows ----------
// dynamic LDS layout:
//   [0, 66560)      : phase 1 raw scores f32[16][1024] (64KB), later reused as V^T bf16[64][520]
//   [66560, 99328)  : blended probs bf16[16][1024] (32KB)
#define VT_STR 520
#define PROB_OFF 66560
__global__ __launch_bounds__(256) void attn_topm_kernel(
    const unsigned short* __restrict__ qbf, const unsigned short* __restrict__ kvbf,
    const float* __restrict__ wblend, unsigned short* __restrict__ obf)
{
    extern __shared__ char smem[];
    float*          raw  = (float*)smem;
    unsigned short* vt   = (unsigned short*)smem;
    unsigned short* prob = (unsigned short*)(smem + PROB_OFF);

    const int tid = threadIdx.x, wid = tid >> 5, lane = tid & 31;
    const int m16 = lane & 15, hf = lane >> 4;
    const int bh = blockIdx.y;
    const int b = bh / HH, h = bh % HH;
    const int q0 = blockIdx.x * 16;

    // ---- Phase 1: raw = q @ k^T (q pre-scaled); fragments are direct b128 global loads
    AFrag aq[2];
    {
        const unsigned short* qrow = qbf + (long)(b * NN + q0 + m16) * DD + h * DHH;
        #pragma unroll
        for (int ks = 0; ks < 2; ++ks) {
            aq[ks].q[0] = *(const uint4*)(qrow + ks * 32 + hf * 8);
            aq[ks].q[1] = *(const uint4*)(qrow + ks * 32 + 16 + hf * 8);
        }
    }
    for (int it = 0; it < 8; ++it) {
        const int kt = wid * 8 + it;                    // 16-key tile
        const unsigned short* krow = kvbf + (long)(b * NN + kt * 16 + m16) * (2 * DD) + h * DHH;
        CFrag c;
        #pragma unroll
        for (int r = 0; r < 8; ++r) c.f[r] = 0.f;
        #pragma unroll
        for (int ks = 0; ks < 2; ++ks) {
            AFrag bfr;                                  // B = K^T, dh in table order (hf*8, 16+hf*8)
            bfr.q[0] = *(const uint4*)(krow + ks * 32 + hf * 8);
            bfr.q[1] = *(const uint4*)(krow + ks * 32 + 16 + hf * 8);
            c.v = __builtin_amdgcn_wmma_f32_16x16x32_bf16(
                false, aq[ks].v, false, bfr.v, (short)0, c.v, false, false);
        }
        #pragma unroll
        for (int r = 0; r < 8; ++r)
            raw[(hf * 8 + r) * 1024 + kt * 16 + m16] = c.f[r];
    }
    __syncthreads();

    // ---- Phase 2: per-row softmaxes (wave owns 2 rows, scores cached in regs)
    float rv0[32], rv1[32];
    const int row0 = wid * 2;
    #pragma unroll
    for (int j = 0; j < 32; ++j) {
        rv0[j] = raw[row0 * 1024 + lane + 32 * j];
        rv1[j] = raw[(row0 + 1) * 1024 + lane + 32 * j];
    }
    const float w0 = wblend[0], w1 = wblend[1];
    const float wm = fmaxf(w0, w1);
    const float e0 = __expf(w0 - wm), e1 = __expf(w1 - wm);
    const float wn0 = e0 / (e0 + e1), wn1 = e1 / (e0 + e1);
    process_row(rv0, prob + row0 * 1024, lane, wn0, wn1);
    process_row(rv1, prob + (row0 + 1) * 1024, lane, wn0, wn1);
    __syncthreads();   // prob ready; raw region now dead -> reuse as V^T

    // ---- Phase 3: o = a @ v in two 512-key halves; V^T staged into LDS
    CFrag c;
    #pragma unroll
    for (int r = 0; r < 8; ++r) c.f[r] = 0.f;
    for (int halfk = 0; halfk < 2; ++halfk) {
        // all 256 threads stage V^T[dh][key] for keys [halfk*512, halfk*512+512)
        for (int i = 0; i < 16; ++i) {
            const int cidx = i * 256 + tid;             // 0..4095
            const int keyl = cidx >> 3;                 // 0..511
            const int dh8  = (cidx & 7) * 8;
            uint4 d = *(const uint4*)(kvbf + (long)(b * NN + halfk * 512 + keyl) * (2 * DD)
                                      + DD + h * DHH + dh8);
            const unsigned short* p = (const unsigned short*)&d;
            #pragma unroll
            for (int j = 0; j < 8; ++j)
                vt[(dh8 + j) * VT_STR + keyl] = p[j];
        }
        __syncthreads();
        if (wid < 4) {                                  // wave = one 16-wide dh tile
            const int nt = wid;
            for (int ki = 0; ki < 16; ++ki) {
                const int k0l = ki * 32;
                const int k0  = halfk * 512 + k0l;
                AFrag a;                                // probs: k in table order (hf*8, 16+hf*8)
                a.q[0] = *(const uint4*)(prob + m16 * 1024 + k0 + hf * 8);
                a.q[1] = *(const uint4*)(prob + m16 * 1024 + k0 + 16 + hf * 8);
                AFrag bfr;                              // V^T: key in table order (hf*8, 16+hf*8)
                bfr.q[0] = *(const uint4*)(&vt[(nt * 16 + m16) * VT_STR + k0l + hf * 8]);
                bfr.q[1] = *(const uint4*)(&vt[(nt * 16 + m16) * VT_STR + k0l + 16 + hf * 8]);
                c.v = __builtin_amdgcn_wmma_f32_16x16x32_bf16(
                    false, a.v, false, bfr.v, (short)0, c.v, false, false);
            }
        }
        __syncthreads();
    }
    if (wid < 4) {
        #pragma unroll
        for (int r = 0; r < 8; ++r)
            obf[((long)(b * NN + q0 + hf * 8 + r)) * DD + h * DHH + wid * 16 + m16] =
                f32_to_bf16_bits(c.f[r]);
    }
}

// ---------- host orchestration ----------
extern "C" void kernel_launch(void* const* d_in, const int* in_sizes, int n_in,
                              void* d_out, int out_size, void* d_ws, size_t ws_size,
                              hipStream_t stream)
{
    (void)in_sizes; (void)n_in; (void)out_size; (void)ws_size;
    const float* x    = (const float*)d_in[0];
    const float* ln1w = (const float*)d_in[1];
    const float* ln1b = (const float*)d_in[2];
    const float* Wq   = (const float*)d_in[3];
    const float* bq   = (const float*)d_in[4];
    const float* Wkv  = (const float*)d_in[5];
    const float* bkv  = (const float*)d_in[6];
    const float* Wo   = (const float*)d_in[7];
    const float* bo   = (const float*)d_in[8];
    const float* wbl  = (const float*)d_in[9];
    const float* ln2w = (const float*)d_in[10];
    const float* ln2b = (const float*)d_in[11];
    const float* W1   = (const float*)d_in[12];
    const float* b1   = (const float*)d_in[13];
    const float* W2   = (const float*)d_in[14];
    const float* b2   = (const float*)d_in[15];

    char* ws = (char*)d_ws;
    size_t off = 0;
    auto alloc = [&](size_t bytes) -> void* {
        void* p = ws + off;
        off += (bytes + 255) & ~(size_t)255;
        return p;
    };
    float*          xbuf  = (float*)         alloc((size_t)BN_TOT * DD * 4);
    unsigned short* hbf   = (unsigned short*)alloc((size_t)BN_TOT * DD * 2);
    unsigned short* qbf   = (unsigned short*)alloc((size_t)BN_TOT * DD * 2);
    unsigned short* kvbf  = (unsigned short*)alloc((size_t)BN_TOT * 2 * DD * 2);
    unsigned short* obf   = (unsigned short*)alloc((size_t)BN_TOT * DD * 2);
    unsigned short* midbf = (unsigned short*)alloc((size_t)BN_TOT * FF * 2);
    unsigned short* WqB   = (unsigned short*)alloc((size_t)LNUM * DD * DD * 2);
    unsigned short* WkvB  = (unsigned short*)alloc((size_t)LNUM * DD * 2 * DD * 2);
    unsigned short* WoB   = (unsigned short*)alloc((size_t)LNUM * DD * DD * 2);
    unsigned short* W1B   = (unsigned short*)alloc((size_t)LNUM * DD * FF * 2);
    unsigned short* W2B   = (unsigned short*)alloc((size_t)LNUM * FF * DD * 2);

    cvt_bf16_kernel<<<2048, 256, 0, stream>>>(Wq,  WqB,  LNUM * DD * DD);
    cvt_bf16_kernel<<<2048, 256, 0, stream>>>(Wkv, WkvB, LNUM * DD * 2 * DD);
    cvt_bf16_kernel<<<2048, 256, 0, stream>>>(Wo,  WoB,  LNUM * DD * DD);
    cvt_bf16_kernel<<<2048, 256, 0, stream>>>(W1,  W1B,  LNUM * DD * FF);
    cvt_bf16_kernel<<<2048, 256, 0, stream>>>(W2,  W2B,  LNUM * FF * DD);
    copy_f32_kernel<<<2048, 256, 0, stream>>>(x, xbuf, BN_TOT * DD);

    const dim3 blk(256);
    const dim3 gQ(DD / 64, BN_TOT / 128);          // N=512
    const dim3 gKV(2 * DD / 64, BN_TOT / 128);     // N=1024
    const dim3 gF(FF / 64, BN_TOT / 128);          // N=2048
    const dim3 gAttn(NN / 16, BB * HH);
    const size_t attnLds = PROB_OFF + 16 * 1024 * sizeof(unsigned short);  // 99328

    for (int l = 0; l < LNUM; ++l) {
        ln_bf16_kernel<<<BN_TOT, blk, 0, stream>>>(xbuf, ln1w + l * DD, ln1b + l * DD, hbf);
        gemm_bf16_kernel<<<gQ, blk, 0, stream>>>(hbf, WqB + (size_t)l * DD * DD,
            bq + l * DD, nullptr, nullptr, qbf, BN_TOT, DD, DD, QSCALE, 0);
        gemm_bf16_kernel<<<gKV, blk, 0, stream>>>(hbf, WkvB + (size_t)l * DD * 2 * DD,
            bkv + l * 2 * DD, nullptr, nullptr, kvbf, BN_TOT, 2 * DD, DD, 1.f, 0);
        attn_topm_kernel<<<gAttn, blk, attnLds, stream>>>(qbf, kvbf, wbl + l * 2, obf);
        gemm_bf16_kernel<<<gQ, blk, 0, stream>>>(obf, WoB + (size_t)l * DD * DD,
            bo + l * DD, xbuf, xbuf, nullptr, BN_TOT, DD, DD, 1.f, 0);
        ln_bf16_kernel<<<BN_TOT, blk, 0, stream>>>(xbuf, ln2w + l * DD, ln2b + l * DD, hbf);
        gemm_bf16_kernel<<<gF, blk, 0, stream>>>(hbf, W1B + (size_t)l * DD * FF,
            b1 + l * FF, nullptr, nullptr, midbf, BN_TOT, FF, DD, 1.f, 1);
        gemm_bf16_kernel<<<gQ, blk, 0, stream>>>(midbf, W2B + (size_t)l * FF * DD,
            b2 + l * DD, xbuf, xbuf, nullptr, BN_TOT, DD, FF, 1.f, 0);
    }
    copy_f32_kernel<<<2048, 256, 0, stream>>>(xbuf, (float*)d_out, BN_TOT * DD);
}
